// BitSelfAttention_84061099917689
// MI455X (gfx1250) — matrix operand order repaired and checked by reading, not compiled, not run on hardware
//
#include <hip/hip_runtime.h>
#include <hip/hip_bf16.h>
#include <stdint.h>

typedef __bf16 bf16;
typedef __attribute__((ext_vector_type(16))) __bf16 v16bf;
typedef __attribute__((ext_vector_type(8)))  float  v8f;
typedef __attribute__((ext_vector_type(4)))  unsigned int u32x4;
typedef __attribute__((ext_vector_type(8)))  int i32x8;
typedef __attribute__((ext_vector_type(4)))  int i32x4;

#if defined(__has_builtin)
#if __has_builtin(__builtin_amdgcn_tensor_load_to_lds) && __has_builtin(__builtin_amdgcn_s_wait_tensorcnt)
#define USE_TDM 1
#endif
#endif
#ifndef USE_TDM
#define USE_TDM 0
#endif

union FragBF { v16bf v; uint4 q[2]; };

static __device__ __forceinline__ bf16 f2bf(float f) {
  uint32_t u = __builtin_bit_cast(uint32_t, f);
  u += 0x7FFFu + ((u >> 16) & 1u);           // round-to-nearest-even
  uint16_t h = (uint16_t)(u >> 16);
  return __builtin_bit_cast(bf16, h);
}
static __device__ __forceinline__ float bf2f(bf16 b) {
  uint32_t u = ((uint32_t)__builtin_bit_cast(uint16_t, b)) << 16;
  return __builtin_bit_cast(float, u);
}

static __device__ __forceinline__ void store_val(float* p, float v) { *p = v; }
static __device__ __forceinline__ void store_val(bf16*  p, float v) { *p = f2bf(v); }

// low 32 bits of a flat shared-aperture address are the LDS byte offset
static __device__ __forceinline__ uint32_t lds_u32(const void* p) {
  return (uint32_t)(uintptr_t)p;
}

#if USE_TDM
// Issue one TDM 2D tile load: tile_x contiguous elements (bf16) per row,
// tile_y rows, row stride = stride_elems. D# per CDNA5 ISA sec 8.
static __device__ __forceinline__ void tdm_load_2d(uint32_t lds_addr, const void* gaddr,
                                                   uint32_t tile_x, uint32_t tile_y,
                                                   uint32_t stride_elems) {
  const uint64_t ga = (uint64_t)(uintptr_t)gaddr;
  u32x4 g0;
  g0[0] = 1u;                                                   // count=1, user descriptor
  g0[1] = lds_addr;                                             // lds_addr[31:0]
  g0[2] = (uint32_t)ga;                                         // global_addr[31:0]
  g0[3] = (uint32_t)((ga >> 32) & 0x01FFFFFFu) | 0x80000000u;   // global_addr[56:32] | type=2
  i32x8 g1;
  g1[0] = (int)(1u << 16);                                      // wg_mask=0, data_size=1 (2B)
  g1[1] = (int)((tile_x & 0xFFFFu) << 16);                      // tensor_dim0[15:0]
  g1[2] = (int)(((tile_x >> 16) & 0xFFFFu) | ((tile_y & 0xFFFFu) << 16)); // dim0 hi | dim1 lo
  g1[3] = (int)(((tile_y >> 16) & 0xFFFFu) | ((tile_x & 0xFFFFu) << 16)); // dim1 hi | tile_dim0
  g1[4] = (int)(tile_y & 0xFFFFu);                              // tile_dim1 | tile_dim2=0
  g1[5] = (int)stride_elems;                                    // tensor_dim0_stride[31:0]
  g1[6] = 0;                                                    // stride0 hi | stride1 lo
  g1[7] = 0;                                                    // stride1 hi
  const i32x4 z4 = {0, 0, 0, 0};
  const i32x8 z8 = {0, 0, 0, 0, 0, 0, 0, 0};
  __builtin_amdgcn_tensor_load_to_lds(g0, g1, z4, z4, z8, 0);
}
#endif

// ---------------------------------------------------------------- fp32 -> bf16
__global__ void k_f32_to_bf16(const float* __restrict__ in, bf16* __restrict__ out, int n) {
  int i = blockIdx.x * blockDim.x + threadIdx.x;
  if (i < n) out[i] = f2bf(in[i]);
}

// ------------------------------------------------- ternary dequant (group=128)
__global__ __launch_bounds__(256) void k_dequant(const float* __restrict__ w, bf16* __restrict__ out) {
  const int wave = threadIdx.x >> 5;
  const int lane = threadIdx.x & 31;
  const int gid  = blockIdx.x * 8 + wave;
  const size_t base = (size_t)gid * 128;
  float4 v = *(const float4*)(w + base + lane * 4);
  float s = fabsf(v.x) + fabsf(v.y) + fabsf(v.z) + fabsf(v.w);
  #pragma unroll
  for (int m = 1; m < 32; m <<= 1) s += __shfl_xor(s, m, 32);
  float scale = fmaxf(s * (1.0f / 128.0f), 1e-8f);
  float inv = 1.0f / scale;
  float q[4] = { v.x * inv, v.y * inv, v.z * inv, v.w * inv };
  bf16* dst = out + base + lane * 4;
  #pragma unroll
  for (int j = 0; j < 4; j++) {
    float t = q[j] > 0.5f ? 1.0f : (q[j] < -0.5f ? -1.0f : 0.0f);
    dst[j] = f2bf(t * scale);
  }
}

// --------------------------------------------------------- GEMM: C = X * W^T
// X: MxK bf16 row-major, W: NxK bf16 row-major. 128x128x32 tiles, 8 waves.
// TDM path: wave 0 DMA's the next K-tile into LDS (double buffered) while all
// waves run WMMAs on the current tile; sync via TENSORcnt + barrier.
template <typename OutT>
__global__ __launch_bounds__(256) void k_gemm_xwT(const bf16* __restrict__ X,
                                                  const bf16* __restrict__ W,
                                                  OutT* __restrict__ C,
                                                  int M, int N, int K) {
#if USE_TDM
  __shared__ __align__(16) bf16 As[2][128][32];
  __shared__ __align__(16) bf16 Bs[2][128][32];
#else
  __shared__ __align__(16) bf16 As[1][128][32];
  __shared__ __align__(16) bf16 Bs[1][128][32];
#endif
  const int tid  = threadIdx.x;
  const int lane = tid & 31, w = tid >> 5;
  const int ln = lane & 15, h = lane >> 4;
  const int wm = w >> 1, wn = w & 1;
  const int bm = blockIdx.y * 128, bn = blockIdx.x * 128;

  v8f acc[2][4];
  #pragma unroll
  for (int a = 0; a < 2; a++)
    #pragma unroll
    for (int b = 0; b < 4; b++)
      #pragma unroll
      for (int e = 0; e < 8; e++) acc[a][b][e] = 0.0f;

#if USE_TDM
  const bf16* agbase = X + (size_t)bm * K;
  const bf16* bgbase = W + (size_t)bn * K;
  if (w == 0) {
    tdm_load_2d(lds_u32(&As[0][0][0]), agbase, 32, 128, (uint32_t)K);
    tdm_load_2d(lds_u32(&Bs[0][0][0]), bgbase, 32, 128, (uint32_t)K);
  }
#else
  const int lrow = tid >> 1;
  const int lk   = (tid & 1) * 16;
  const bf16* xrow = X + (size_t)(bm + lrow) * K + lk;
  const bf16* wrow = W + (size_t)(bn + lrow) * K + lk;
#endif

  for (int kt = 0; kt < K; kt += 32) {
#if USE_TDM
    const int cur = (kt >> 5) & 1;
    if (w == 0) {
      if (kt + 32 < K) {
        tdm_load_2d(lds_u32(&As[cur ^ 1][0][0]), agbase + kt + 32, 32, 128, (uint32_t)K);
        tdm_load_2d(lds_u32(&Bs[cur ^ 1][0][0]), bgbase + kt + 32, 32, 128, (uint32_t)K);
        __builtin_amdgcn_s_wait_tensorcnt(2);  // 2 newest outstanding -> current tile landed
      } else {
        __builtin_amdgcn_s_wait_tensorcnt(0);
      }
    }
    __syncthreads();
#else
    const int cur = 0;
    *(uint4*)&As[0][lrow][lk]     = *(const uint4*)(xrow + kt);
    *(uint4*)&As[0][lrow][lk + 8] = *(const uint4*)(xrow + kt + 8);
    *(uint4*)&Bs[0][lrow][lk]     = *(const uint4*)(wrow + kt);
    *(uint4*)&Bs[0][lrow][lk + 8] = *(const uint4*)(wrow + kt + 8);
    __syncthreads();
#endif

    FragBF a[2];
    #pragma unroll
    for (int mi = 0; mi < 2; mi++) {
      const int m = wm * 32 + mi * 16 + ln;
      a[mi].q[0] = *(const uint4*)&As[cur][m][h * 8];
      a[mi].q[1] = *(const uint4*)&As[cur][m][16 + h * 8];
    }
    #pragma unroll
    for (int ni = 0; ni < 4; ni++) {
      FragBF bfr;
      const int n = wn * 64 + ni * 16 + ln;
      bfr.q[0] = *(const uint4*)&Bs[cur][n][h * 8];
      bfr.q[1] = *(const uint4*)&Bs[cur][n][16 + h * 8];
      #pragma unroll
      for (int mi = 0; mi < 2; mi++)
        acc[mi][ni] = __builtin_amdgcn_wmma_f32_16x16x32_bf16(
            false, a[mi].v, false, bfr.v, (short)0, acc[mi][ni], false, false);
    }
    __syncthreads();
  }

  #pragma unroll
  for (int mi = 0; mi < 2; mi++)
    #pragma unroll
    for (int ni = 0; ni < 4; ni++)
      #pragma unroll
      for (int i = 0; i < 8; i++) {
        const int row = bm + wm * 32 + mi * 16 + i + 8 * h;
        const int col = bn + wn * 64 + ni * 16 + ln;
        store_val(C + (size_t)row * N + col, acc[mi][ni][i]);
      }
}

// ------------------------------------------------------------------- RoPE
__global__ void k_rope(const bf16* __restrict__ in, bf16* __restrict__ out, int T) {
  const int t = blockIdx.x, hd = blockIdx.y, b = blockIdx.z;
  const int j = threadIdx.x;  // 0..63
  const float invf = powf(10000.0f, -(float)j * (1.0f / 64.0f));
  const float fr = (float)t * invf;
  const float c = cosf(fr), s = sinf(fr);
  const bf16* src = in + ((size_t)(b * T + t)) * 2048 + hd * 128;
  const float x1 = bf2f(src[j]), x2 = bf2f(src[j + 64]);
  bf16* dst = out + (((size_t)(b * 16 + hd)) * T + t) * 128;
  dst[j]      = f2bf(x1 * c - x2 * s);
  dst[j + 64] = f2bf(x1 * s + x2 * c);
}

// -------------------------------------------- V transpose: (B,T,2048)->(B,H,128,T)
__global__ void k_vtrans(const bf16* __restrict__ in, bf16* __restrict__ out, int T) {
  const int i = blockIdx.x * blockDim.x + threadIdx.x;  // over B*16*128*T
  const int t = i % T;
  const int d = (i / T) & 127;
  const int hb = i / (T * 128);
  const int hd = hb & 15, b = hb >> 4;
  out[i] = in[((size_t)(b * T + t)) * 2048 + hd * 128 + d];
}

// ------------------------------------------------------ flash attention (causal)
__global__ __launch_bounds__(128) void k_attn(const bf16* __restrict__ Q,
                                              const bf16* __restrict__ Km,
                                              const bf16* __restrict__ VT,
                                              bf16* __restrict__ Y, int T) {
  __shared__ __align__(16) bf16 plds[4][16 * 32];
  const int lane = threadIdx.x & 31, w = threadIdx.x >> 5;
  const int ln = lane & 15, h = lane >> 4;
  const int b = blockIdx.z, hd = blockIdx.y;
  const int qbase = blockIdx.x * 64 + w * 16;
  const bf16* qp = Q  + ((size_t)(b * 16 + hd) * T) * 128;
  const bf16* kp = Km + ((size_t)(b * 16 + hd) * T) * 128;
  const bf16* vp = VT + ((size_t)(b * 16 + hd) * 128) * T;

  FragBF aq[4];
  {
    const bf16* qr = qp + (size_t)(qbase + ln) * 128;
    #pragma unroll
    for (int c = 0; c < 4; c++) {
      aq[c].q[0] = *(const uint4*)(qr + c * 32 + h * 8);
      aq[c].q[1] = *(const uint4*)(qr + c * 32 + 16 + h * 8);
    }
  }
  v8f O[8];
  float mi_[8], li_[8];
  #pragma unroll
  for (int t = 0; t < 8; t++)
    #pragma unroll
    for (int e = 0; e < 8; e++) O[t][e] = 0.0f;
  #pragma unroll
  for (int e = 0; e < 8; e++) { mi_[e] = -3.0e38f; li_[e] = 0.0f; }

  const float sc = 0.08838834764831845f;  // 1/sqrt(128)
  const int nkt = (qbase + 16 + 31) >> 5; // causal tile bound, per wave
  bf16* pl = plds[w];

  for (int kt = 0; kt < nkt; kt++) {
    const int kb = kt * 32;
    v8f S0, S1;
    #pragma unroll
    for (int e = 0; e < 8; e++) { S0[e] = 0.0f; S1[e] = 0.0f; }
    #pragma unroll
    for (int c = 0; c < 4; c++) {
      FragBF kf;
      const bf16* kr0 = kp + (size_t)(kb + ln) * 128 + c * 32 + h * 8;
      kf.q[0] = *(const uint4*)(kr0);
      kf.q[1] = *(const uint4*)(kr0 + 16);
      S0 = __builtin_amdgcn_wmma_f32_16x16x32_bf16(false, aq[c].v, false, kf.v, (short)0, S0, false, false);
      const bf16* kr1 = kp + (size_t)(kb + 16 + ln) * 128 + c * 32 + h * 8;
      kf.q[0] = *(const uint4*)(kr1);
      kf.q[1] = *(const uint4*)(kr1 + 16);
      S1 = __builtin_amdgcn_wmma_f32_16x16x32_bf16(false, aq[c].v, false, kf.v, (short)0, S1, false, false);
    }
    #pragma unroll
    for (int i = 0; i < 8; i++) {
      const int qrow = qbase + i + 8 * h;
      float s0 = S0[i] * sc;
      float s1 = S1[i] * sc;
      if (kb + ln > qrow)      s0 = -3.0e38f;
      if (kb + 16 + ln > qrow) s1 = -3.0e38f;
      float mx = fmaxf(s0, s1);
      mx = fmaxf(mx, __shfl_xor(mx, 1, 32));
      mx = fmaxf(mx, __shfl_xor(mx, 2, 32));
      mx = fmaxf(mx, __shfl_xor(mx, 4, 32));
      mx = fmaxf(mx, __shfl_xor(mx, 8, 32));
      const float nm = fmaxf(mi_[i], mx);
      const float p0 = __expf(s0 - nm);
      const float p1 = __expf(s1 - nm);
      float rs = p0 + p1;
      rs += __shfl_xor(rs, 1, 32);
      rs += __shfl_xor(rs, 2, 32);
      rs += __shfl_xor(rs, 4, 32);
      rs += __shfl_xor(rs, 8, 32);
      const float alpha = __expf(mi_[i] - nm);
      li_[i] = li_[i] * alpha + rs;
      mi_[i] = nm;
      #pragma unroll
      for (int t = 0; t < 8; t++) O[t][i] *= alpha;
      pl[(i + 8 * h) * 32 + ln]      = f2bf(p0);   // C-layout -> LDS
      pl[(i + 8 * h) * 32 + 16 + ln] = f2bf(p1);
    }
    __asm__ volatile("" ::: "memory");
    __builtin_amdgcn_wave_barrier();   // same-wave DS ops are in-order
    FragBF pa;                          // reload P in A-layout
    pa.q[0] = *(const uint4*)&pl[ln * 32 + h * 8];
    pa.q[1] = *(const uint4*)&pl[ln * 32 + 16 + h * 8];
    __asm__ volatile("" ::: "memory");
    __builtin_amdgcn_wave_barrier();
    #pragma unroll
    for (int t = 0; t < 8; t++) {
      FragBF vf;
      const bf16* vr = vp + (size_t)(t * 16 + ln) * T + kb + h * 8;
      vf.q[0] = *(const uint4*)(vr);
      vf.q[1] = *(const uint4*)(vr + 16);
      O[t] = __builtin_amdgcn_wmma_f32_16x16x32_bf16(false, pa.v, false, vf.v, (short)0, O[t], false, false);
    }
  }

  #pragma unroll
  for (int i = 0; i < 8; i++) {
    const int row = qbase + i + 8 * h;
    const float rinv = 1.0f / li_[i];
    bf16* yr = Y + ((size_t)(b * T) + row) * 2048 + hd * 128;
    #pragma unroll
    for (int t = 0; t < 8; t++)
      yr[t * 16 + ln] = f2bf(O[t][i] * rinv);
  }
}

// --------------------------------------------------------------------- launch
extern "C" void kernel_launch(void* const* d_in, const int* in_sizes, int n_in,
                              void* d_out, int out_size, void* d_ws, size_t ws_size,
                              hipStream_t stream) {
  const float* x  = (const float*)d_in[0];
  const float* wq = (const float*)d_in[1];
  const float* wk = (const float*)d_in[2];
  const float* wv = (const float*)d_in[3];
  const float* wo = (const float*)d_in[4];
  float* out = (float*)d_out;

  const int Bb = 2, Tt = 2048, D = 2048, H = 16;
  const int M = Bb * Tt;                // 4096
  char* ws = (char*)d_ws;
  const size_t MB = 1024ull * 1024ull;
  bf16* xb  = (bf16*)(ws);              // 16MB  x in bf16
  bf16* wqb = (bf16*)(ws + 16 * MB);
  bf16* wkb = (bf16*)(ws + 24 * MB);
  bf16* wvb = (bf16*)(ws + 32 * MB);
  bf16* wob = (bf16*)(ws + 40 * MB);
  bf16* tmp = (bf16*)(ws + 48 * MB);    // 16MB projection scratch
  bf16* qh  = (bf16*)(ws + 64 * MB);    // (B,H,T,128)
  bf16* kh  = (bf16*)(ws + 80 * MB);
  bf16* vt  = (bf16*)(ws + 96 * MB);    // (B,H,128,T)
  bf16* yb  = (bf16*)(ws + 112 * MB);   // attention out (B,T,2048)

  k_f32_to_bf16<<<(M * D + 255) / 256, 256, 0, stream>>>(x, xb, M * D);
  k_dequant<<<(D * 16) / 8, 256, 0, stream>>>(wq, wqb);
  k_dequant<<<(D * 16) / 8, 256, 0, stream>>>(wk, wkb);
  k_dequant<<<(D * 16) / 8, 256, 0, stream>>>(wv, wvb);
  k_dequant<<<(D * 16) / 8, 256, 0, stream>>>(wo, wob);

  dim3 gg(D / 128, M / 128);
  dim3 gr(Tt, H, Bb);
  k_gemm_xwT<bf16><<<gg, 256, 0, stream>>>(xb, wqb, tmp, M, D, D);
  k_rope<<<gr, 64, 0, stream>>>(tmp, qh, Tt);
  k_gemm_xwT<bf16><<<gg, 256, 0, stream>>>(xb, wkb, tmp, M, D, D);
  k_rope<<<gr, 64, 0, stream>>>(tmp, kh, Tt);
  k_gemm_xwT<bf16><<<gg, 256, 0, stream>>>(xb, wvb, tmp, M, D, D);
  k_vtrans<<<(Bb * H * 128 * Tt + 255) / 256, 256, 0, stream>>>(tmp, vt, Tt);

  dim3 ga(Tt / 64, H, Bb);
  k_attn<<<ga, 128, 0, stream>>>(qh, kh, vt, yb, Tt);

  k_gemm_xwT<float><<<gg, 256, 0, stream>>>(yb, wob, out, M, D, D);
}
